// GAT_90254442758204
// MI455X (gfx1250) — matrix operand order repaired and checked by reading, not compiled, not run on hardware
//
#include <hip/hip_runtime.h>
#include <stdint.h>

// ---------- types ----------
typedef __bf16 v16bf __attribute__((ext_vector_type(16)));
typedef float  v8f   __attribute__((ext_vector_type(8)));
typedef unsigned int uint4v __attribute__((ext_vector_type(4)));
typedef int    i4v   __attribute__((ext_vector_type(4)));

#define NEG_SLOPE 0.2f

// ---------- async global->LDS support (gfx1250) ----------
#if defined(__has_builtin)
#  if __has_builtin(__builtin_amdgcn_global_load_async_to_lds_b128)
#    define USE_ASYNC_LDS 1
#  endif
#endif
#ifndef USE_ASYNC_LDS
#  define USE_ASYNC_LDS 0
#endif

static __device__ __forceinline__ void copy16_g2l(const unsigned short* g,
                                                  unsigned short* l) {
#if USE_ASYNC_LDS
  __builtin_amdgcn_global_load_async_to_lds_b128(
      (__attribute__((address_space(1))) i4v*)g,
      (__attribute__((address_space(3))) i4v*)l, 0, 0);
#else
  *(uint4v*)l = *(const uint4v*)g;
#endif
}

static __device__ __forceinline__ void async_copy_wait() {
#if USE_ASYNC_LDS
#  if defined(__has_builtin) && __has_builtin(__builtin_amdgcn_s_wait_asynccnt)
  __builtin_amdgcn_s_wait_asynccnt(0);
#  else
  asm volatile("s_wait_asynccnt 0" ::: "memory");
#  endif
#endif
}

static __device__ __forceinline__ unsigned short f32_to_bf16_rne(float f) {
  unsigned u = __float_as_uint(f);
  unsigned r = 0x7FFFu + ((u >> 16) & 1u);
  u += r;
  return (unsigned short)(u >> 16);
}

// monotone-float atomic max (works for mixed signs; init with -3.4e38f)
static __device__ __forceinline__ void atomicMaxFloat(float* addr, float v) {
  if (v >= 0.f) atomicMax((int*)addr, __float_as_int(v));
  else          atomicMin((unsigned int*)addr, __float_as_uint(v));
}

// ---------- elementwise helpers ----------
__global__ void k_f32_to_bf16(const float* __restrict__ in,
                              unsigned short* __restrict__ out, int n) {
  int i = blockIdx.x * blockDim.x + threadIdx.x;
  if (i < n) out[i] = f32_to_bf16_rne(in[i]);
}

// out[n*K + k] = bf16(in[k*256 + n])  -- transpose [K x 256] -> [256 x K]
__global__ void k_convT_bf16(const float* __restrict__ in,
                             unsigned short* __restrict__ out, int K) {
  int i = blockIdx.x * blockDim.x + threadIdx.x;
  if (i >= K * 256) return;
  int n = i / K, k = i % K;
  out[i] = f32_to_bf16_rne(in[(size_t)k * 256 + n]);
}

__global__ void k_fill_f32(float* __restrict__ p, float v, int n) {
  int i = blockIdx.x * blockDim.x + threadIdx.x;
  if (i < n) p[i] = v;
}

// ---------- WMMA GEMM: C[rows x 256] = A[rows x K] * B[K x 256] ----------
// A: bf16 row-major [rows x K]; Bt: bf16 TRANSPOSED [256 x K]; C: f32 [rows x 256]
// rows % 16 == 0, K % 64 == 0.
// grid = (rows/16, 2), block = 128 (4 waves). Wave w covers cols
// blockIdx.y*128 + w*32 .. +31 (two 16x16 tiles, shared A fragment).
__global__ __launch_bounds__(128) void k_wmma_gemm_256(
    const unsigned short* __restrict__ A, const unsigned short* __restrict__ Bt,
    float* __restrict__ C, int K) {
  __shared__ __align__(16) unsigned short ldsA[16 * 64];     // 2 KB
  __shared__ __align__(16) unsigned short ldsBT[128 * 64];   // 16 KB

  const int t    = threadIdx.x;
  const int lane = t & 31;
  const int wave = t >> 5;
  const int row0 = blockIdx.x * 16;
  const int col0 = blockIdx.y * 128;
  const int hi   = (lane >> 4) & 1;   // lanes 16-31 -> 1
  const int l15  = lane & 15;

  v8f acc0 = {0.f, 0.f, 0.f, 0.f, 0.f, 0.f, 0.f, 0.f};
  v8f acc1 = {0.f, 0.f, 0.f, 0.f, 0.f, 0.f, 0.f, 0.f};

  const int koff = hi ? 8 : 0;    // A fragment K offset
  const int kb   = hi ? 8 : 0;    // B fragment K offset

  for (int kk = 0; kk < K; kk += 64) {
    // stage A tile 16x64 bf16 (128 chunks of 16B; one per thread)
    {
      int r = t >> 3, seg = t & 7;
      copy16_g2l(&A[(size_t)(row0 + r) * K + kk + seg * 8],
                 &ldsA[r * 64 + seg * 8]);
    }
    // stage B^T slab: 128 cols x 64 k bf16 (1024 chunks; 8 per thread)
#pragma unroll
    for (int i = 0; i < 8; ++i) {
      int c = i * 128 + t;
      int n = c >> 3, seg = c & 7;
      copy16_g2l(&Bt[(size_t)(col0 + n) * K + kk + seg * 8],
                 &ldsBT[n * 64 + seg * 8]);
    }
    async_copy_wait();
    __syncthreads();

#pragma unroll
    for (int s = 0; s < 2; ++s) {
      union { uint4v q[2]; v16bf v; } ua, ub0, ub1;
      // A fragment: row M=l15; elems 0-7 -> K=koff.., 8-15 -> K=koff+16..
      const unsigned short* pa = &ldsA[l15 * 64 + s * 32 + koff];
      ua.q[0] = *(const uint4v*)(pa);
      ua.q[1] = *(const uint4v*)(pa + 16);
      // B fragments: lane col N=l15 of each 16-wide tile; elems 0-7 -> K=kb.., 8-15 -> K=kb+16..
      const unsigned short* pb0 = &ldsBT[(wave * 32 + l15) * 64 + s * 32 + kb];
      ub0.q[0] = *(const uint4v*)(pb0);
      ub0.q[1] = *(const uint4v*)(pb0 + 16);
      const unsigned short* pb1 = pb0 + 16 * 64;
      ub1.q[0] = *(const uint4v*)(pb1);
      ub1.q[1] = *(const uint4v*)(pb1 + 16);

      acc0 = __builtin_amdgcn_wmma_f32_16x16x32_bf16(false, ua.v, false, ub0.v,
                                                     (short)0, acc0, false, false);
      acc1 = __builtin_amdgcn_wmma_f32_16x16x32_bf16(false, ua.v, false, ub1.v,
                                                     (short)0, acc1, false, false);
    }
    __syncthreads();
  }

  // C layout: VGPR r, lane -> M = r + hi*8, N = l15
  const int cb0 = col0 + wave * 32 + l15;
#pragma unroll
  for (int r = 0; r < 8; ++r) {
    size_t rowoff = (size_t)(row0 + r + hi * 8) * 256;
    C[rowoff + cb0]      = acc0[r];
    C[rowoff + cb0 + 16] = acc1[r];
  }
}

// ---------- attention logits: al[n,h] = sum_c hlin[n, h*C + c] * a[h*C + c] ----------
__global__ void k_logits(const float* __restrict__ hlin, const float* __restrict__ a,
                         float* __restrict__ al, int Nn, int H, int Cc) {
  int i = blockIdx.x * blockDim.x + threadIdx.x;
  if (i >= Nn * H) return;
  int n = i / H, h = i % H;
  const float* hp = hlin + (size_t)n * (H * Cc) + h * Cc;
  const float* ap = a + h * Cc;
  float s = 0.f;
  for (int c = 0; c < Cc; ++c) s += hp[c] * ap[c];
  al[i] = s;
}

// ---------- edge pass 1: segment max over dst ----------
__global__ void k_edge_max(const int* __restrict__ ei, int E, int Etot,
                           const float* __restrict__ als, const float* __restrict__ ald,
                           float* __restrict__ m, int H) {
  int i = blockIdx.x * blockDim.x + threadIdx.x;
  if (i >= Etot * H) return;
  int e = i / H, h = i % H;
  int s = (e < E) ? ei[e] : (e - E);
  int d = (e < E) ? ei[E + e] : (e - E);
  float v = als[s * H + h] + ald[d * H + h];
  v = (v >= 0.f) ? v : NEG_SLOPE * v;
  atomicMaxFloat(&m[d * H + h], v);
}

// ---------- edge pass 2: exp(e - max) and segment sum ----------
__global__ void k_edge_expsum(const int* __restrict__ ei, int E, int Etot,
                              const float* __restrict__ als, const float* __restrict__ ald,
                              const float* __restrict__ m, float* __restrict__ sbuf,
                              float* __restrict__ abuf, int H) {
  int i = blockIdx.x * blockDim.x + threadIdx.x;
  if (i >= Etot * H) return;
  int e = i / H, h = i % H;
  int s = (e < E) ? ei[e] : (e - E);
  int d = (e < E) ? ei[E + e] : (e - E);
  float v = als[s * H + h] + ald[d * H + h];
  v = (v >= 0.f) ? v : NEG_SLOPE * v;
  float ex = __expf(v - m[d * H + h]);
  abuf[i] = ex;
  atomicAdd(&sbuf[d * H + h], ex);
}

// ---------- edge pass 3: out[dst, :] += alpha * hlin[src, :] (one block per edge) ----------
__global__ __launch_bounds__(256) void k_edge_scatter(
    const int* __restrict__ ei, int E,
    const float* __restrict__ hlin, const float* __restrict__ sbuf,
    const float* __restrict__ abuf, float* __restrict__ out, int H) {
  int e = blockIdx.x;
  int j = threadIdx.x;           // 0..255
  int s = (e < E) ? ei[e] : (e - E);
  int d = (e < E) ? ei[E + e] : (e - E);
  int Cc = 256 / H;
  int h = j / Cc;
  float alpha = abuf[(size_t)e * H + h] / sbuf[d * H + h];
  atomicAdd(&out[(size_t)d * 256 + j], alpha * hlin[(size_t)s * 256 + j]);
}

// ---------- bias + relu + convert to bf16 ----------
__global__ void k_bias_relu_bf16(const float* __restrict__ acc, const float* __restrict__ b,
                                 unsigned short* __restrict__ out, int total) {
  int i = blockIdx.x * blockDim.x + threadIdx.x;
  if (i >= total) return;
  float v = acc[i] + b[i & 255];
  v = v > 0.f ? v : 0.f;
  out[i] = f32_to_bf16_rne(v);
}

// ---------- pooling ----------
__global__ void k_pool(const float* __restrict__ acc, const float* __restrict__ b2,
                       const int* __restrict__ batch, float* __restrict__ psum, int Nn) {
  int i = blockIdx.x * blockDim.x + threadIdx.x;
  if (i >= Nn * 256) return;
  int n = i >> 8, j = i & 255;
  atomicAdd(&psum[(size_t)batch[n] * 256 + j], acc[i] + b2[j]);
}

__global__ void k_count(const int* __restrict__ batch, float* __restrict__ cnt, int Nn) {
  int n = blockIdx.x * blockDim.x + threadIdx.x;
  if (n < Nn) atomicAdd(&cnt[batch[n]], 1.0f);
}

// ---------- final linear: out[g,l] = (psum[g,:]/max(cnt,1)) . lin_w[:,l] + lin_b[l] ----------
__global__ void k_final(const float* __restrict__ psum, const float* __restrict__ cnt,
                        const float* __restrict__ lw, const float* __restrict__ lb,
                        float* __restrict__ out) {
  int t = blockIdx.x * blockDim.x + threadIdx.x;
  if (t >= 64 * 10) return;
  int g = t / 10, l = t % 10;
  float c = cnt[g]; c = (c > 1.f) ? c : 1.f;
  float s = 0.f;
  for (int j = 0; j < 256; ++j) s += psum[(size_t)g * 256 + j] * lw[j * 10 + l];
  out[t] = s / c + lb[l];
}

// ---------- host launch ----------
extern "C" void kernel_launch(void* const* d_in, const int* in_sizes, int n_in,
                              void* d_out, int out_size, void* d_ws, size_t ws_size,
                              hipStream_t stream) {
  const float* x      = (const float*)d_in[0];
  const int*   ei     = (const int*)d_in[1];
  const int*   batch  = (const int*)d_in[2];
  const float* W1     = (const float*)d_in[3];
  const float* a_src1 = (const float*)d_in[4];
  const float* a_dst1 = (const float*)d_in[5];
  const float* b1     = (const float*)d_in[6];
  const float* W2     = (const float*)d_in[7];
  const float* a_src2 = (const float*)d_in[8];
  const float* a_dst2 = (const float*)d_in[9];
  const float* b2     = (const float*)d_in[10];
  const float* lw     = (const float*)d_in[11];
  const float* lb     = (const float*)d_in[12];
  float* out = (float*)d_out;

  const int N    = in_sizes[2];          // 10000 (multiple of 16)
  const int E    = in_sizes[1] / 2;      // 320000
  const int Etot = E + N;                // self-loops appended
  const int FIN  = in_sizes[0] / N;      // 128
  const int H1   = 16;

  // workspace carve-out (aligned 256B)
  char* ws = (char*)d_ws;
  size_t off = 0;
  auto carve = [&](size_t bytes) -> void* {
    void* p = ws + off;
    off = (off + bytes + 255) & ~(size_t)255;
    return p;
  };
  unsigned short* xb   = (unsigned short*)carve((size_t)N * FIN * 2);
  unsigned short* W1t  = (unsigned short*)carve((size_t)FIN * 256 * 2);  // [256 x FIN]
  float* hlin          = (float*)carve((size_t)N * 256 * 4);   // h1lin, later h2lin
  float* acc           = (float*)carve((size_t)N * 256 * 4);   // acc1, later acc2
  unsigned short* h1b  = (unsigned short*)carve((size_t)N * 256 * 2);
  unsigned short* W2t  = (unsigned short*)carve((size_t)256 * 256 * 2);  // [256 x 256]
  float* als           = (float*)carve((size_t)N * H1 * 4);
  float* ald           = (float*)carve((size_t)N * H1 * 4);
  float* mbuf          = (float*)carve((size_t)N * H1 * 4);
  float* sbuf          = (float*)carve((size_t)N * H1 * 4);
  float* abuf          = (float*)carve((size_t)Etot * H1 * 4);
  float* psum          = (float*)carve((size_t)64 * 256 * 4);
  float* cnt           = (float*)carve((size_t)64 * 4);
  (void)ws_size; (void)n_in; (void)out_size;

  const int TB = 256;
  auto blocks = [&](long n) { return (int)((n + TB - 1) / TB); };

  // ---- layer 1 ----
  k_f32_to_bf16<<<blocks((long)N * FIN), TB, 0, stream>>>(x, xb, N * FIN);
  k_convT_bf16<<<blocks((long)FIN * 256), TB, 0, stream>>>(W1, W1t, FIN);

  dim3 g1(N / 16, 2);
  k_wmma_gemm_256<<<g1, 128, 0, stream>>>(xb, W1t, hlin, FIN);

  k_logits<<<blocks((long)N * H1), TB, 0, stream>>>(hlin, a_src1, als, N, H1, 16);
  k_logits<<<blocks((long)N * H1), TB, 0, stream>>>(hlin, a_dst1, ald, N, H1, 16);

  k_fill_f32<<<blocks((long)N * H1), TB, 0, stream>>>(mbuf, -3.402823e38f, N * H1);
  (void)hipMemsetAsync(sbuf, 0, (size_t)N * H1 * 4, stream);
  (void)hipMemsetAsync(acc, 0, (size_t)N * 256 * 4, stream);

  k_edge_max<<<blocks((long)Etot * H1), TB, 0, stream>>>(ei, E, Etot, als, ald, mbuf, H1);
  k_edge_expsum<<<blocks((long)Etot * H1), TB, 0, stream>>>(ei, E, Etot, als, ald, mbuf, sbuf, abuf, H1);
  k_edge_scatter<<<Etot, 256, 0, stream>>>(ei, E, hlin, sbuf, abuf, acc, H1);

  k_bias_relu_bf16<<<blocks((long)N * 256), TB, 0, stream>>>(acc, b1, h1b, N * 256);

  // ---- layer 2 ----
  k_convT_bf16<<<blocks((long)256 * 256), TB, 0, stream>>>(W2, W2t, 256);
  k_wmma_gemm_256<<<g1, 128, 0, stream>>>(h1b, W2t, hlin, 256);  // hlin now = h2lin

  k_logits<<<blocks((long)N), TB, 0, stream>>>(hlin, a_src2, als, N, 1, 256);
  k_logits<<<blocks((long)N), TB, 0, stream>>>(hlin, a_dst2, ald, N, 1, 256);

  k_fill_f32<<<blocks((long)N), TB, 0, stream>>>(mbuf, -3.402823e38f, N);
  (void)hipMemsetAsync(sbuf, 0, (size_t)N * 4, stream);
  (void)hipMemsetAsync(acc, 0, (size_t)N * 256 * 4, stream);

  k_edge_max<<<blocks((long)Etot), TB, 0, stream>>>(ei, E, Etot, als, ald, mbuf, 1);
  k_edge_expsum<<<blocks((long)Etot), TB, 0, stream>>>(ei, E, Etot, als, ald, mbuf, sbuf, abuf, 1);
  k_edge_scatter<<<Etot, 256, 0, stream>>>(ei, E, hlin, sbuf, abuf, acc, 1);

  // ---- pool + classify ----
  (void)hipMemsetAsync(psum, 0, (size_t)64 * 256 * 4, stream);
  (void)hipMemsetAsync(cnt, 0, (size_t)64 * 4, stream);
  k_pool<<<blocks((long)N * 256), TB, 0, stream>>>(acc, b2, batch, psum, N);
  k_count<<<blocks((long)N), TB, 0, stream>>>(batch, cnt, N);
  k_final<<<3, 256, 0, stream>>>(psum, cnt, lw, lb, out);
}
